// MultiheadAttention_55980603736454
// MI455X (gfx1250) — matrix operand order repaired and checked by reading, not compiled, not run on hardware
//
#include <hip/hip_runtime.h>
#include <hip/hip_bf16.h>

#define SEQ   4096
#define BATCH 2
#define EMB   512
#define HEADS 8
#define HD    64

typedef __attribute__((ext_vector_type(16))) _Float16 v16h;
typedef __attribute__((ext_vector_type(8)))  _Float16 v8h;
typedef __attribute__((ext_vector_type(8)))  float    v8f;
typedef __attribute__((ext_vector_type(4)))  float    f32x4;

union HU { v16h v; v8h h[2]; };

__device__ __forceinline__ v8f wmma_f16(v16h a, v16h b, v8f c) {
    return __builtin_amdgcn_wmma_f32_16x16x32_f16(false, a, false, b, (short)0, c, false, false);
}

// convert 8 contiguous f32 -> 8 f16 (p must be 16B aligned)
__device__ __forceinline__ v8h cvt8(const float* __restrict__ p) {
    f32x4 a = *(const f32x4*)p;
    f32x4 b = *(const f32x4*)(p + 4);
    v8h r;
#pragma unroll
    for (int i = 0; i < 4; ++i) { r[i] = (_Float16)a[i]; r[4 + i] = (_Float16)b[i]; }
    return r;
}

// ---------------- Wo f32 -> f16 ----------------
__global__ void cvt_kernel(const float* __restrict__ in, _Float16* __restrict__ out, int n) {
    int i = blockIdx.x * blockDim.x + threadIdx.x;
    if (i < n) out[i] = (_Float16)in[i];
}

// ---------------- fused QKV projection: y = (x @ W^T + b) [*scale] ----------------
// x: [S*B, E] f32 (row m = s*B+b), W: [E,E] f32 row-major (out,in), bias [E].
// mode 0: Q -> f16 [h][b][s][d], value scaled by 1/sqrt(D) (bias included, per reference)
// mode 1: K -> f16 [h][b][s][d]
// mode 2: V -> f16 [h][b][d][s]   (transposed so s is K-contiguous for P@V B-tiles)
__global__ __launch_bounds__(256) void proj_kernel(const float* __restrict__ x,
                                                   const float* __restrict__ w,
                                                   const float* __restrict__ bias,
                                                   _Float16* __restrict__ out, int mode) {
    const int lane = threadIdx.x & 31;
    const int wave = threadIdx.x >> 5;
    const int l16  = lane & 15;
    const int hlf  = lane >> 4;

    const int tilesN = EMB / 32;                 // 16
    int tile = blockIdx.x * 8 + wave;            // 4096 tiles total
    int tm = tile / tilesN, tn = tile % tilesN;
    int m0 = tm * 32, n0 = tn * 32;

    v8f acc[2][2] = {};
    for (int k0 = 0; k0 < EMB; k0 += 32) {
        v16h A[2], Bm[2];
#pragma unroll
        for (int i = 0; i < 2; ++i) {
            const float* ap = x + (size_t)(m0 + i * 16 + l16) * EMB + k0;
            HU u;
            u.h[0] = cvt8(ap + hlf * 8);
            u.h[1] = cvt8(ap + 16 + hlf * 8);
            A[i] = u.v;
        }
#pragma unroll
        for (int j = 0; j < 2; ++j) {
            const float* bp = w + (size_t)(n0 + j * 16 + l16) * EMB + k0 + hlf * 8;
            HU u;
            u.h[0] = cvt8(bp);
            u.h[1] = cvt8(bp + 16);
            Bm[j] = u.v;
        }
#pragma unroll
        for (int i = 0; i < 2; ++i)
#pragma unroll
            for (int j = 0; j < 2; ++j)
                acc[i][j] = wmma_f16(A[i], Bm[j], acc[i][j]);
    }

    const float qscale = (mode == 0) ? 0.125f : 1.0f;  // 1/sqrt(64)
#pragma unroll
    for (int j = 0; j < 2; ++j) {
        int n = n0 + j * 16 + l16;
        float bn = bias[n];
        int hh = n >> 6, d = n & 63;
#pragma unroll
        for (int i = 0; i < 2; ++i) {
#pragma unroll
            for (int r = 0; r < 8; ++r) {
                int m = m0 + i * 16 + hlf * 8 + r;
                int s = m >> 1, b = m & 1;
                float v = (acc[i][j][r] + bn) * qscale;
                size_t off;
                if (mode == 2) off = ((size_t)((hh * BATCH + b) * HD) + d) * SEQ + s;
                else           off = ((size_t)((hh * BATCH + b) * SEQ) + s) * HD + d;
                out[off] = (_Float16)v;
            }
        }
    }
}

// ---------------- flash attention: one wave handles 16 query rows of one (h,b) ----------------
// qh/kh: [h][b][s][d] f16 (q pre-scaled), vt: [h][b][d][s] f16.
// X out: [s][h*128 + b*64 + d] f16  == reference's reshape(S,B,E) view, i.e. [8192][512] row-major.
__global__ __launch_bounds__(32) void attn_kernel(const _Float16* __restrict__ qh,
                                                  const _Float16* __restrict__ kh,
                                                  const _Float16* __restrict__ vt,
                                                  _Float16* __restrict__ X) {
    const int lane = threadIdx.x & 31;
    const int l16 = lane & 15, hlf = lane >> 4;

    const int qtiles = SEQ / 16;                 // 256
    int id = blockIdx.x;
    int qt = id % qtiles;
    int hb = id / qtiles;
    int hh = hb / BATCH, bb = hb % BATCH;
    const _Float16* qp = qh + (size_t)hb * SEQ * HD;
    const _Float16* kp = kh + (size_t)hb * SEQ * HD;
    const _Float16* vp = vt + (size_t)hb * HD * SEQ;
    int q0 = qt * 16;

    __shared__ __align__(16) _Float16 Plds[2][16 * 32];

    // Q A-tiles: 16 rows x 64 (K-dim d), two 16x32 tiles, loaded once
    v16h QA[2];
#pragma unroll
    for (int t = 0; t < 2; ++t) {
        const _Float16* p = qp + (size_t)(q0 + l16) * HD + t * 32;
        HU u;
        u.h[0] = *(const v8h*)(p + hlf * 8);
        u.h[1] = *(const v8h*)(p + 16 + hlf * 8);
        QA[t] = u.v;
    }

    float mrow[8], lrow[8];
#pragma unroll
    for (int r = 0; r < 8; ++r) { mrow[r] = -1e30f; lrow[r] = 0.0f; }
    v8f O[4] = {};

    for (int kb = 0; kb < SEQ; kb += 32) {
        int buf = (kb >> 5) & 1;

        // ---- S = Q @ K^T for 32 keys: two 16x16 f32 tiles ----
        v8f Sv[2];
#pragma unroll
        for (int j = 0; j < 2; ++j) {
            const _Float16* bp = kp + (size_t)(kb + j * 16 + l16) * HD;
            HU b0, b1;
            b0.h[0] = *(const v8h*)(bp + hlf * 8);
            b0.h[1] = *(const v8h*)(bp + 16 + hlf * 8);
            b1.h[0] = *(const v8h*)(bp + 32 + hlf * 8);
            b1.h[1] = *(const v8h*)(bp + 32 + 16 + hlf * 8);
            v8f z = {};
            v8f s = wmma_f16(QA[0], b0.v, z);
            Sv[j] = wmma_f16(QA[1], b1.v, s);
        }

        // ---- online softmax (rows live on groups of 16 lanes; reduce via shfl_xor 1/2/4/8) ----
#pragma unroll
        for (int r = 0; r < 8; ++r) {
            float mx = fmaxf(Sv[0][r], Sv[1][r]);
#pragma unroll
            for (int off = 1; off < 16; off <<= 1)
                mx = fmaxf(mx, __shfl_xor(mx, off, 32));
            float mnew  = fmaxf(mrow[r], mx);
            float alpha = __expf(mrow[r] - mnew);
            mrow[r] = mnew;
            float p0 = __expf(Sv[0][r] - mnew);
            float p1 = __expf(Sv[1][r] - mnew);
            float rs = p0 + p1;
#pragma unroll
            for (int off = 1; off < 16; off <<= 1)
                rs += __shfl_xor(rs, off, 32);
            lrow[r] = lrow[r] * alpha + rs;
#pragma unroll
            for (int t = 0; t < 4; ++t) O[t][r] *= alpha;
            int row = hlf * 8 + r;
            Plds[buf][row * 32 + l16]      = (_Float16)p0;
            Plds[buf][row * 32 + 16 + l16] = (_Float16)p1;
        }
        __syncthreads();  // single-wave WG: compiler memory fence for the C->A relayout

        // ---- P (A-layout from LDS) @ V (B-tiles from transposed V) ----
        HU pa;
        pa.h[0] = *(const v8h*)&Plds[buf][l16 * 32 + hlf * 8];
        pa.h[1] = *(const v8h*)&Plds[buf][l16 * 32 + 16 + hlf * 8];
#pragma unroll
        for (int t = 0; t < 4; ++t) {
            const _Float16* vpp = vp + (size_t)(t * 16 + l16) * SEQ + kb + hlf * 8;
            HU u;
            u.h[0] = *(const v8h*)vpp;
            u.h[1] = *(const v8h*)(vpp + 16);
            O[t] = wmma_f16(pa.v, u.v, O[t]);
        }

        if (kb + 32 < SEQ) {  // -> global_prefetch_b8 for next chunk
            __builtin_prefetch(kp + (size_t)(kb + 32 + l16) * HD, 0, 1);
            __builtin_prefetch(vp + (size_t)l16 * SEQ + kb + 32, 0, 1);
        }
    }

    // ---- normalize + store in the reference's reshape layout ----
#pragma unroll
    for (int r = 0; r < 8; ++r) {
        float inv = 1.0f / lrow[r];
        int s = q0 + hlf * 8 + r;
        size_t base = (size_t)s * (BATCH * EMB) + hh * 128 + bb * 64 + l16;
#pragma unroll
        for (int t = 0; t < 4; ++t)
            X[base + t * 16] = (_Float16)(O[t][r] * inv);
    }
}

// ---------------- output projection: out = X @ Wo^T + bo (f32 out) ----------------
__global__ __launch_bounds__(256) void outproj_kernel(const _Float16* __restrict__ X,
                                                      const _Float16* __restrict__ wo,
                                                      const float* __restrict__ bo,
                                                      float* __restrict__ out) {
    const int lane = threadIdx.x & 31;
    const int wave = threadIdx.x >> 5;
    const int l16 = lane & 15, hlf = lane >> 4;

    const int tilesN = EMB / 32;
    int tile = blockIdx.x * 8 + wave;
    int tm = tile / tilesN, tn = tile % tilesN;
    int m0 = tm * 32, n0 = tn * 32;

    v8f acc[2][2] = {};
    for (int k0 = 0; k0 < EMB; k0 += 32) {
        v16h A[2], Bm[2];
#pragma unroll
        for (int i = 0; i < 2; ++i) {
            const _Float16* ap = X + (size_t)(m0 + i * 16 + l16) * EMB + k0;
            HU u;
            u.h[0] = *(const v8h*)(ap + hlf * 8);
            u.h[1] = *(const v8h*)(ap + 16 + hlf * 8);
            A[i] = u.v;
        }
#pragma unroll
        for (int j = 0; j < 2; ++j) {
            const _Float16* bp = wo + (size_t)(n0 + j * 16 + l16) * EMB + k0 + hlf * 8;
            HU u;
            u.h[0] = *(const v8h*)bp;
            u.h[1] = *(const v8h*)(bp + 16);
            Bm[j] = u.v;
        }
#pragma unroll
        for (int i = 0; i < 2; ++i)
#pragma unroll
            for (int j = 0; j < 2; ++j)
                acc[i][j] = wmma_f16(A[i], Bm[j], acc[i][j]);
    }

#pragma unroll
    for (int j = 0; j < 2; ++j) {
        int n = n0 + j * 16 + l16;
        float bn = bo[n];
#pragma unroll
        for (int i = 0; i < 2; ++i)
#pragma unroll
            for (int r = 0; r < 8; ++r) {
                int m = m0 + i * 16 + hlf * 8 + r;
                out[(size_t)m * EMB + n] = acc[i][j][r] + bn;
            }
    }
}

extern "C" void kernel_launch(void* const* d_in, const int* in_sizes, int n_in,
                              void* d_out, int out_size, void* d_ws, size_t ws_size,
                              hipStream_t stream) {
    const float* query = (const float*)d_in[0];
    const float* key   = (const float*)d_in[1];
    const float* value = (const float*)d_in[2];
    const float* Wq = (const float*)d_in[3];
    const float* bq = (const float*)d_in[4];
    const float* Wk = (const float*)d_in[5];
    const float* bk = (const float*)d_in[6];
    const float* Wv = (const float*)d_in[7];
    const float* bv = (const float*)d_in[8];
    const float* Wo = (const float*)d_in[9];
    const float* bo = (const float*)d_in[10];

    const size_t QKV_BYTES = (size_t)HEADS * BATCH * SEQ * HD * sizeof(_Float16);  // 8 MB
    char* ws = (char*)d_ws;
    _Float16* qh   = (_Float16*)(ws + 0 * QKV_BYTES);
    _Float16* khb  = (_Float16*)(ws + 1 * QKV_BYTES);
    _Float16* vtb  = (_Float16*)(ws + 2 * QKV_BYTES);
    _Float16* Xb   = (_Float16*)(ws + 3 * QKV_BYTES);   // [S][B*E] f16, 8 MB
    _Float16* wo16 = (_Float16*)(ws + 4 * QKV_BYTES);   // 0.5 MB

    cvt_kernel<<<(EMB * EMB + 255) / 256, 256, 0, stream>>>(Wo, wo16, EMB * EMB);

    const int gemm_blocks = (SEQ * BATCH / 32) * (EMB / 32) / 8;  // 512
    proj_kernel<<<gemm_blocks, 256, 0, stream>>>(query, Wq, bq, qh, 0);
    proj_kernel<<<gemm_blocks, 256, 0, stream>>>(key,   Wk, bk, khb, 1);
    proj_kernel<<<gemm_blocks, 256, 0, stream>>>(value, Wv, bv, vtb, 2);

    attn_kernel<<<HEADS * BATCH * (SEQ / 16), 32, 0, stream>>>(qh, khb, vtb, Xb);

    outproj_kernel<<<gemm_blocks, 256, 0, stream>>>(Xb, wo16, bo, (float*)d_out);
}
